// SSMTimeSeriesModel_83373905150196
// MI455X (gfx1250) — matrix operand order repaired and checked by reading, not compiled, not run on hardware
//
#include <hip/hip_runtime.h>

// SSM scan on MI455X (gfx1250, wave32, WMMA).
// k1: convert A (512x512) and Wb (512x32) to bf16 (L2-resident operands).
// k2: persistent recurrent scan, 8 WGs x 16 batch rows, state in LDS,
//     v_wmma_f32_16x16x32_bf16 for s@A^T with fused x_t@Wb^T tail + LN + ReLU.
//     B-operand (A matrix) streams from L2 each step through an opaque,
//     explicitly global (AS1) pointer: defeats LICM (no scratch spills) while
//     emitting global_load_b128 (LOADcnt only, no flat/DScnt contention).
// k3: decoder GEMM [128,512]x[512,16384] with v_wmma_f32_16x16x4_f32 (fp32 Wd).

typedef __attribute__((ext_vector_type(8)))  float  v8f;
typedef __attribute__((ext_vector_type(2)))  float  v2f;
typedef __attribute__((ext_vector_type(16))) __bf16 v16bf;
typedef __attribute__((ext_vector_type(8)))  __bf16 v8bf;

#define GLOBAL_AS __attribute__((address_space(1)))
typedef GLOBAL_AS const unsigned short* gus_p;
typedef GLOBAL_AS const v8bf*           gv8_p;

#define BSZ 128
#define SSZ 512
#define CIN 32
#define DD  512
#define SB_STRIDE 520   // bf16 state row stride (halfs), 16B aligned
#define SF_STRIDE 516   // fp32 staging row stride (floats)

union BfFrag { v16bf v; v8bf h[2]; };

__device__ __forceinline__ unsigned short f2bf(float f) {
    unsigned int u = __float_as_uint(f);
    u += 0x7FFFu + ((u >> 16) & 1u);      // round-to-nearest-even
    return (unsigned short)(u >> 16);
}

// ---------------- k1: bf16 conversion of A and Wb ----------------
__global__ void cvt_bf16_kernel(const float* __restrict__ A,
                                const float* __restrict__ Wb,
                                unsigned short* __restrict__ Abf,
                                unsigned short* __restrict__ Wbbf) {
    int i = blockIdx.x * 256 + threadIdx.x;
    if (i < DD * DD) Abf[i]  = f2bf(A[i]);
    if (i < DD * CIN) Wbbf[i] = f2bf(Wb[i]);
}

// ---------------- k2: recurrent scan ----------------
// grid = 8 (16 batch rows each), block = 1024 (32 waves; wave w owns N-tile w).
__global__ __launch_bounds__(1024) void ssm_scan_kernel(
    const float* __restrict__ x,            // [B,S,CIN]
    const unsigned short* __restrict__ Abf, // [D,D] bf16 row-major (B-operand: k contiguous)
    const unsigned short* __restrict__ Wbbf,// [D,CIN] bf16
    const float* __restrict__ bb,
    const float* __restrict__ lng,
    const float* __restrict__ lnb,
    float* __restrict__ fs)                 // [B,D] final state fp32
{
    __shared__ unsigned short Sb[16 * SB_STRIDE]; // state, bf16 (A-operand source)
    __shared__ float          Sf[16 * SF_STRIDE]; // pre-LN accumulators, fp32
    __shared__ unsigned short Xb[2][16 * CIN];    // x_t tile, bf16, double buffered
    __shared__ float Ps[64 * 16];                 // LN partial sums
    __shared__ float Pq[64 * 16];                 // LN partial sums of squares
    __shared__ float BBl[DD], Gl[DD], Bl[DD];
    __shared__ float Mu[16], Rs[16];

    const int tid  = threadIdx.x;
    const int lane = tid & 31;
    const int wave = tid >> 5;      // 0..31 == N tile index
    const int m16  = lane & 15;
    const int hi   = lane >> 4;     // 0/1 lane half
    const int b0   = blockIdx.x * 16;
    const int colL = wave * 16 + m16;   // column owned by this lane (B-frag / C store)

    // zero initial state, preload LN params, stage x_0
    for (int i = tid; i < 16 * DD; i += 1024) {
        int r = i >> 9, c = i & 511;
        Sb[r * SB_STRIDE + c] = 0;
    }
    if (tid < DD) { BBl[tid] = bb[tid]; Gl[tid] = lng[tid]; Bl[tid] = lnb[tid]; }
    if (tid < 16 * CIN) {
        int r = tid >> 5, k = tid & 31;
        Xb[0][r * CIN + k] = f2bf(x[((size_t)(b0 + r) * SSZ + 0) * CIN + k]);
    }
    __syncthreads();

    const int rrow = tid & 15;      // LN role: row
    const int rg   = tid >> 4;      // LN role: group 0..63

    // per-thread register copies of loop-invariant params
    const float bbv = BBl[colL];
    float gv[8], bvv[8];
    #pragma unroll
    for (int j = 0; j < 8; ++j) { gv[j] = Gl[rg + 64 * j]; bvv[j] = Bl[rg + 64 * j]; }

    for (int t = 0; t < SSZ; ++t) {
        // prefetch x_{t+1} into the other buffer (overlaps with WMMA k-loop;
        // made visible by the end-of-step barrier)
        if (t + 1 < SSZ && tid < 16 * CIN) {
            int r = tid >> 5, k = tid & 31;
            Xb[(t + 1) & 1][r * CIN + k] =
                f2bf(x[((size_t)(b0 + r) * SSZ + (t + 1)) * CIN + k]);
        }

        // Opaque global base: defeats LICM (B streams from L2 each step) while
        // keeping address space 1 so loads lower to global_load_b128.
        unsigned long long abase = (unsigned long long)Abf;
        asm volatile("" : "+s"(abase));
        gus_p AbG = (gus_p)abase;

        v8f acc0 = {}, acc1 = {};
        // s_{t-1} @ A^T : K = 512, two interleaved accumulator chains
        for (int k0 = 0; k0 < DD; k0 += 64) {
            BfFrag a0, b0, a1, b1;
            a0.h[0] = *(const v8bf*)&Sb[m16 * SB_STRIDE + k0 + hi * 8];
            a0.h[1] = *(const v8bf*)&Sb[m16 * SB_STRIDE + k0 + 16 + hi * 8];
            gus_p bp0 = AbG + (size_t)colL * DD + k0 + hi * 8;
            b0.h[0] = *(gv8_p)bp0;
            b0.h[1] = *(gv8_p)(bp0 + 16);
            acc0 = __builtin_amdgcn_wmma_f32_16x16x32_bf16(
                false, a0.v, false, b0.v, (short)0, acc0, false, false);

            a1.h[0] = *(const v8bf*)&Sb[m16 * SB_STRIDE + k0 + 32 + hi * 8];
            a1.h[1] = *(const v8bf*)&Sb[m16 * SB_STRIDE + k0 + 48 + hi * 8];
            gus_p bp1 = bp0 + 32;
            b1.h[0] = *(gv8_p)bp1;
            b1.h[1] = *(gv8_p)(bp1 + 16);
            acc1 = __builtin_amdgcn_wmma_f32_16x16x32_bf16(
                false, a1.v, false, b1.v, (short)0, acc1, false, false);
        }
        // fused Bu tail: x_t @ Wb^T (K = 32, one WMMA; Wb frag stays hoisted)
        {
            BfFrag a, b;
            a.h[0] = *(const v8bf*)&Xb[t & 1][m16 * CIN + hi * 8];
            a.h[1] = *(const v8bf*)&Xb[t & 1][m16 * CIN + 16 + hi * 8];
            const unsigned short* bp = Wbbf + (size_t)colL * CIN + hi * 8;
            b.h[0] = *(const v8bf*)bp;
            b.h[1] = *(const v8bf*)(bp + 16);
            acc0 = __builtin_amdgcn_wmma_f32_16x16x32_bf16(
                false, a.v, false, b.v, (short)0, acc0, false, false);
        }
        // C fragment -> Sf (+ bb bias); VGPR i holds row i+8*hi, col = lane%16
        #pragma unroll
        for (int i = 0; i < 8; ++i)
            Sf[(i + hi * 8) * SF_STRIDE + colL] = acc0[i] + acc1[i] + bbv;
        __syncthreads();

        // LN stage A: 64 groups x 16 rows, each thread sums 8 elems
        {
            float s = 0.f, q = 0.f;
            #pragma unroll
            for (int j = 0; j < 8; ++j) {
                float v = Sf[rrow * SF_STRIDE + rg + 64 * j];
                s += v; q += v * v;
            }
            Ps[rg * 16 + rrow] = s;
            Pq[rg * 16 + rrow] = q;
        }
        __syncthreads();
        // LN stage B: 16 waves, one row each; butterfly reduce across 32 lanes
        if (tid < 512) {
            int w = wave;  // row 0..15
            float s = Ps[lane * 16 + w] + Ps[(lane + 32) * 16 + w];
            float q = Pq[lane * 16 + w] + Pq[(lane + 32) * 16 + w];
            #pragma unroll
            for (int off = 16; off > 0; off >>= 1) {
                s += __shfl_xor(s, off, 32);
                q += __shfl_xor(q, off, 32);
            }
            if (lane == 0) {
                float mu  = s * (1.f / 512.f);
                float var = q * (1.f / 512.f) - mu * mu;
                Mu[w] = mu;
                Rs[w] = rsqrtf(var + 1e-5f);
            }
        }
        __syncthreads();
        // apply LN + ReLU, write bf16 state for next step (fp32 out on last step)
        {
            float mu = Mu[rrow], rs = Rs[rrow];
            #pragma unroll
            for (int j = 0; j < 8; ++j) {
                int col = rg + 64 * j;
                float v = (Sf[rrow * SF_STRIDE + col] - mu) * rs * gv[j] + bvv[j];
                v = fmaxf(v, 0.f);
                Sb[rrow * SB_STRIDE + col] = f2bf(v);
                if (t == SSZ - 1) fs[(size_t)(b0 + rrow) * DD + col] = v;
            }
        }
        __syncthreads();
    }
}

// ---------------- k3: decoder GEMM, fp32 WMMA ----------------
// out[m][n] = sum_k fs[m][k] * Wd[n][k] + bd[n];  M=128, N=16384, K=512
__global__ __launch_bounds__(256) void ssm_decode_kernel(
    const float* __restrict__ fs,   // [128, 512]
    const float* __restrict__ Wd,   // [16384, 512]
    const float* __restrict__ bd,   // [16384]
    float* __restrict__ out)        // [128, 16384]
{
    const int tid  = threadIdx.x;
    const int lane = tid & 31;
    const int wave = tid >> 5;
    const int m16  = lane & 15;
    const int hi   = lane >> 4;
    const int tile = blockIdx.x * 8 + wave;  // 0..8191
    const int mt   = tile & 7;               // 8 M tiles
    const int nt   = tile >> 3;              // 1024 N tiles

    const float* ap = fs + (size_t)(mt * 16 + m16) * DD;
    const float* bp = Wd + (size_t)(nt * 16 + m16) * DD;

    v8f acc0 = {}, acc1 = {}, acc2 = {}, acc3 = {};
    for (int k0 = 0; k0 < DD; k0 += 16) {
        // fp32 A frag: lane<16 -> (k0,k0+1); lane>=16 -> (k0+2,k0+3)
        v2f a0 = *(const v2f*)(ap + k0 + 2 * hi);
        v2f b0 = *(const v2f*)(bp + k0 + 2 * hi);
        acc0 = __builtin_amdgcn_wmma_f32_16x16x4_f32(
            false, a0, false, b0, (short)0, acc0, false, false);
        v2f a1 = *(const v2f*)(ap + k0 + 4 + 2 * hi);
        v2f b1 = *(const v2f*)(bp + k0 + 4 + 2 * hi);
        acc1 = __builtin_amdgcn_wmma_f32_16x16x4_f32(
            false, a1, false, b1, (short)0, acc1, false, false);
        v2f a2 = *(const v2f*)(ap + k0 + 8 + 2 * hi);
        v2f b2 = *(const v2f*)(bp + k0 + 8 + 2 * hi);
        acc2 = __builtin_amdgcn_wmma_f32_16x16x4_f32(
            false, a2, false, b2, (short)0, acc2, false, false);
        v2f a3 = *(const v2f*)(ap + k0 + 12 + 2 * hi);
        v2f b3 = *(const v2f*)(bp + k0 + 12 + 2 * hi);
        acc3 = __builtin_amdgcn_wmma_f32_16x16x4_f32(
            false, a3, false, b3, (short)0, acc3, false, false);
    }
    const int col = nt * 16 + m16;
    const float bdv = bd[col];
    #pragma unroll
    for (int i = 0; i < 8; ++i) {
        int row = mt * 16 + i + hi * 8;
        out[(size_t)row * 16384 + col] = acc0[i] + acc1[i] + acc2[i] + acc3[i] + bdv;
    }
}

extern "C" void kernel_launch(void* const* d_in, const int* in_sizes, int n_in,
                              void* d_out, int out_size, void* d_ws, size_t ws_size,
                              hipStream_t stream) {
    const float* x   = (const float*)d_in[0];
    const float* A   = (const float*)d_in[1];
    const float* Wb  = (const float*)d_in[2];
    const float* bb  = (const float*)d_in[3];
    const float* lng = (const float*)d_in[4];
    const float* lnb = (const float*)d_in[5];
    const float* Wd  = (const float*)d_in[6];
    const float* bd  = (const float*)d_in[7];
    float* out = (float*)d_out;

    char* ws = (char*)d_ws;
    unsigned short* Abf  = (unsigned short*)ws;                       // 524288 B
    unsigned short* Wbbf = (unsigned short*)(ws + 524288);            //  32768 B
    float*          fs   = (float*)(ws + 524288 + 32768);             // 262144 B

    cvt_bf16_kernel<<<(DD * DD + 255) / 256, 256, 0, stream>>>(A, Wb, Abf, Wbbf);
    ssm_scan_kernel<<<BSZ / 16, 1024, 0, stream>>>(x, Abf, Wbbf, bb, lng, lnb, fs);
    ssm_decode_kernel<<<(8 * 1024) / 8, 256, 0, stream>>>(fs, Wd, bd, out);
}
